// Generator_16492674417364
// MI455X (gfx1250) — matrix operand order repaired and checked by reading, not compiled, not run on hardware
//
#include <hip/hip_runtime.h>
#include <hip/hip_bf16.h>
#include <cstdint>

typedef __bf16 bf16;
typedef __attribute__((ext_vector_type(16))) __bf16 v16bf;
typedef __attribute__((ext_vector_type(8)))  __bf16 v8bf;
typedef __attribute__((ext_vector_type(8)))  float  v8f;
typedef __attribute__((ext_vector_type(4)))  int    v4i;

#define B_ 256
#define S_ 512
#define H_ 512

// workspace layout (bytes)
#define OFF_WBF   0ull                          // 2 * 2048 * 512 bf16 = 4 MiB
#define OFF_HBF   (4ull << 20)                  // 2 * 256 * 512 bf16  = 512 KiB
#define OFF_HF32  (OFF_HBF + (512ull << 10))    // 2 * 256 * 512 f32   = 1 MiB
#define OFF_C     (OFF_HF32 + (1ull << 20))     // 2 * 256 * 512 f32   = 1 MiB
#define OFF_IMP   (OFF_C + (1ull << 20))        // 513 * 2 * 256 f32   = ~1.05 MiB
#define ZERO_OFF  OFF_HBF
#define ZERO_BYTES ((512ull << 10) + (2ull << 20) + 513ull * 512ull * 4ull)

// ---------- CDNA5 async global->LDS copy (ASYNCcnt path), with fallback ----------
#if __has_builtin(__builtin_amdgcn_global_load_async_to_lds_b128)
#define HAVE_ASYNC 1
#else
#define HAVE_ASYNC 0
#endif

__device__ __forceinline__ void wait_async0() {
#if __has_builtin(__builtin_amdgcn_s_wait_asynccnt)
    __builtin_amdgcn_s_wait_asynccnt(0);
#else
    asm volatile("s_wait_asynccnt 0x0" ::: "memory");
#endif
}

#if HAVE_ASYNC
// copy 16 bytes per lane, global -> LDS, tracked by ASYNCcnt
__device__ __forceinline__ void async_cp16(const void* g, void* l) {
    __builtin_amdgcn_global_load_async_to_lds_b128(
        (__attribute__((address_space(1))) v4i*)(uintptr_t)g,
        (__attribute__((address_space(3))) v4i*)(uint32_t)(uintptr_t)l,
        0, 0);
}
#endif

__global__ __launch_bounds__(256) void convert_whh(const float* __restrict__ Wf,
                                                   const float* __restrict__ Wb,
                                                   bf16* __restrict__ out) {
    const int i = blockIdx.x * 256 + threadIdx.x;     // 0 .. 2*2048*512-1
    const int N = 2048 * 512;
    const float v = (i < N) ? Wf[i] : Wb[i - N];
    out[i] = (bf16)v;
}

__device__ __forceinline__ float sigm(float x) { return 1.0f / (1.0f + __expf(-x)); }

// All 5 fragment loads issued before the 4 WMMAs -> single dscnt wait per slice.
__device__ __forceinline__ void compute_slice(const bf16* Ab, const bf16* Bb,
                                              int mi, int j, int hl, int ln,
                                              v8f& acc0, v8f& acc1, v8f& acc2, v8f& acc3) {
    const v8bf alo = *(const v8bf*)&Ab[(16 * mi + ln) * 32 + 16 * hl];
    const v8bf ahi = *(const v8bf*)&Ab[(16 * mi + ln) * 32 + 16 * hl + 8];
    const v16bf b0 = *(const v16bf*)&Bb[(0 * 32 + 16 * j + ln) * 32 + 16 * hl];
    const v16bf b1 = *(const v16bf*)&Bb[(1 * 32 + 16 * j + ln) * 32 + 16 * hl];
    const v16bf b2 = *(const v16bf*)&Bb[(2 * 32 + 16 * j + ln) * 32 + 16 * hl];
    const v16bf b3 = *(const v16bf*)&Bb[(3 * 32 + 16 * j + ln) * 32 + 16 * hl];
    v16bf a;
#pragma unroll
    for (int e = 0; e < 8; ++e) { a[e] = alo[e]; a[e + 8] = ahi[e]; }
    acc0 = __builtin_amdgcn_wmma_f32_16x16x32_bf16(false, a, false, b0, (short)0, acc0, false, false);
    acc1 = __builtin_amdgcn_wmma_f32_16x16x32_bf16(false, a, false, b1, (short)0, acc1, false, false);
    acc2 = __builtin_amdgcn_wmma_f32_16x16x32_bf16(false, a, false, b2, (short)0, acc2, false, false);
    acc3 = __builtin_amdgcn_wmma_f32_16x16x32_bf16(false, a, false, b3, (short)0, acc3, false, false);
}

// One time step of both LSTM directions.
// grid = 128 WGs: dir(2) x btile(4, 64 rows) x ntile(16, 32 hidden units)
// block = 256 threads = 8 waves; wave w -> mi = w>>1 (16-row M tile), j = w&1 (16-col N half)
// Each wave computes all 4 gate tiles (i,f,g,o) for its 16x16 (b,n) patch.
__global__ __launch_bounds__(256) void lstm_step(
    const bf16*  __restrict__ Wbf,   // [dir][4H rows][H k] bf16
    bf16*        __restrict__ hbf,   // [dir][B][H] bf16
    float*       __restrict__ hf32,  // [dir][B][H] f32
    float*       __restrict__ cst,   // [dir][B][H] f32
    float*       __restrict__ imp,   // [513][dir][B] f32 (pre-bias imputation dot)
    const float* __restrict__ values, const float* __restrict__ masks,
    const float* __restrict__ Wih_f, const float* __restrict__ bih_f, const float* __restrict__ bhh_f,
    const float* __restrict__ Wih_b, const float* __restrict__ bih_b, const float* __restrict__ bhh_b,
    const float* __restrict__ W_imp, const float* __restrict__ b_imp,
    int t)                            // -1 = initial cell (x = +/-128, h = c = 0)
{
    // depth-3 pipeline: one barrier per K-slice is sufficient
    __shared__ bf16  ldsB[3][128 * 32];  // 4 gates x 32 n rows, 32-k slice (8 KiB each)
    __shared__ bf16  ldsA[3][64 * 32];   // 64 b rows, 32-k slice (4 KiB each)
    __shared__ float cc_s[64];

    const int bx  = blockIdx.x;
    const int dir = bx & 1;
    const int b0  = ((bx >> 1) & 3) * 64;
    const int n0  = (bx >> 3) * 32;

    const float* Wih = dir ? Wih_b : Wih_f;
    const float* bih = dir ? bih_b : bih_f;
    const float* bhh = dir ? bhh_b : bhh_f;
    const bf16*  Wd  = Wbf + (size_t)dir * 2048 * 512;
    const bf16*  Hd  = hbf + (size_t)dir * 256 * 512;

    const int tid = threadIdx.x;

    // Per-step cell input cc[b] = (1-m)*x + m*(imp + b_imp), or +/-128 at init.
    if (tid < 64) {
        const int b = b0 + tid;
        float ccv;
        if (t < 0) {
            ccv = dir ? -128.0f : 128.0f;
        } else {
            const float m  = masks [b * S_ + t];
            const float x  = values[b * S_ + t];
            const float iv = imp[(size_t)t * 512 + dir * 256 + b] + b_imp[0];
            ccv = (1.0f - m) * x + m * iv;
        }
        cc_s[tid] = ccv;
    }

    const int wave = tid >> 5, lane = tid & 31;
    const int mi = wave >> 1, j = wave & 1;
    const int hl = lane >> 4, ln = lane & 15;

    v8f acc0 = {}; v8f acc1 = {}; v8f acc2 = {}; v8f acc3 = {};

    // cooperative staging indices
    const int   srowB  = tid >> 1;              // 0..127
    const int   shB    = (tid & 1) << 4;        // 0 / 16 (k chunk, elements)
    const int   gBr    = srowB >> 5, nlB = srowB & 31;
    const bf16* gB_ptr = Wd + (size_t)(gBr * H_ + n0 + nlB) * H_ + shB;
    const int   lBo    = srowB * 32 + shB;

    const int   srowA  = tid >> 2;              // 0..63
    const int   chA    = (tid & 3) << 3;        // 0,8,16,24 (elements)
    const bf16* gA_ptr = Hd + (size_t)(b0 + srowA) * H_ + chA;
    const int   lAo    = srowA * 32 + chA;

    // ---- stage slice 0 into buffer 0 ----
#if HAVE_ASYNC
    async_cp16(gB_ptr,     &ldsB[0][lBo]);
    async_cp16(gB_ptr + 8, &ldsB[0][lBo + 8]);
    async_cp16(gA_ptr,     &ldsA[0][lAo]);
    wait_async0();
#else
    *(v16bf*)&ldsB[0][lBo] = *(const v16bf*)gB_ptr;
    *(v8bf*)&ldsA[0][lAo]  = *(const v8bf*)gA_ptr;
#endif
    __syncthreads();

#pragma unroll
    for (int i = 0; i < 16; ++i) {
        const int kb  = i << 5;
        const int cur = i % 3;
        const int nxt = (i + 1) % 3;

        // issue prefetch of slice i+1 (lands in a buffer last read at iter i-2)
#if HAVE_ASYNC
        if (i < 15) {
            async_cp16(gB_ptr + kb + 32,     &ldsB[nxt][lBo]);
            async_cp16(gB_ptr + kb + 32 + 8, &ldsB[nxt][lBo + 8]);
            async_cp16(gA_ptr + kb + 32,     &ldsA[nxt][lAo]);
        }
#else
        v16bf rB; v8bf rA;
        if (i < 15) {
            rB = *(const v16bf*)(gB_ptr + kb + 32);
            rA = *(const v8bf*)(gA_ptr + kb + 32);
        }
#endif

        // compute on slice i while the prefetch is in flight
        compute_slice(ldsA[cur], ldsB[cur], mi, j, hl, ln, acc0, acc1, acc2, acc3);

#if HAVE_ASYNC
        if (i < 15) wait_async0();
#else
        if (i < 15) {
            *(v16bf*)&ldsB[nxt][lBo] = rB;
            *(v8bf*)&ldsA[nxt][lAo]  = rA;
        }
#endif
        __syncthreads();   // publish slice i+1; retire reads of slice i
    }

    // Gate fusion + state update. C/D layout: element r -> M = r + 8*hl, N = ln.
    const int   n      = n0 + 16 * j + ln;
    const float wimp_n = W_imp[n];
    const float bs0 = bih[0 * H_ + n] + bhh[0 * H_ + n], wi0 = Wih[0 * H_ + n];
    const float bs1 = bih[1 * H_ + n] + bhh[1 * H_ + n], wi1 = Wih[1 * H_ + n];
    const float bs2 = bih[2 * H_ + n] + bhh[2 * H_ + n], wi2 = Wih[2 * H_ + n];
    const float bs3 = bih[3 * H_ + n] + bhh[3 * H_ + n], wi3 = Wih[3 * H_ + n];

#pragma unroll
    for (int r = 0; r < 8; ++r) {
        const int   brow = 16 * mi + 8 * hl + r;   // 0..63
        const int   b    = b0 + brow;
        const float ccv  = cc_s[brow];

        const float gi = acc0[r] + bs0 + ccv * wi0;
        const float gf = acc1[r] + bs1 + ccv * wi1;
        const float gg = acc2[r] + bs2 + ccv * wi2;
        const float go = acc3[r] + bs3 + ccv * wi3;

        const float i_ = sigm(gi);
        const float f_ = sigm(gf);
        const float g_ = tanhf(gg);
        const float o_ = sigm(go);

        const size_t idx = (size_t)dir * B_ * H_ + (size_t)b * H_ + n;
        const float  cn  = f_ * cst[idx] + i_ * g_;
        const float  hn  = o_ * tanhf(cn);
        cst[idx]  = cn;
        hf32[idx] = hn;
        hbf[idx]  = (bf16)hn;

        // partial imp for next step: reduce 16 n-lanes (same b within a half)
        float p = hn * wimp_n;
        p += __shfl_xor(p, 8, 16);
        p += __shfl_xor(p, 4, 16);
        p += __shfl_xor(p, 2, 16);
        p += __shfl_xor(p, 1, 16);
        if (ln == 0)
            atomicAdd(&imp[(size_t)(t + 1) * 512 + dir * 256 + b], p);
    }
}

__global__ __launch_bounds__(256) void lstm_epilogue(
    const float* __restrict__ hf32, const float* __restrict__ values,
    const float* __restrict__ masks, float* __restrict__ out) {
    const int i = blockIdx.x * 256 + threadIdx.x;      // 0 .. 131071 (b*512+s)
    const float h = hf32[i] + hf32[131072 + i];        // h_f + h_b
    out[i] = h;                                        // first output: h
    const float m = masks[i], x = values[i];
    out[131072 + i] = h * (1.0f - m) + x * m;          // second output: out
}

extern "C" void kernel_launch(void* const* d_in, const int* in_sizes, int n_in,
                              void* d_out, int out_size, void* d_ws, size_t ws_size,
                              hipStream_t stream) {
    const float* values = (const float*)d_in[0];
    const float* masks  = (const float*)d_in[1];
    const float* Wih_f  = (const float*)d_in[2];
    const float* Whh_f  = (const float*)d_in[3];
    const float* bih_f  = (const float*)d_in[4];
    const float* bhh_f  = (const float*)d_in[5];
    const float* Wih_b  = (const float*)d_in[6];
    const float* Whh_b  = (const float*)d_in[7];
    const float* bih_b  = (const float*)d_in[8];
    const float* bhh_b  = (const float*)d_in[9];
    const float* W_imp  = (const float*)d_in[10];
    const float* b_imp  = (const float*)d_in[11];

    char*  ws   = (char*)d_ws;
    bf16*  Wbf  = (bf16*)(ws + OFF_WBF);
    bf16*  hbf  = (bf16*)(ws + OFF_HBF);
    float* hf32 = (float*)(ws + OFF_HF32);
    float* cst  = (float*)(ws + OFF_C);
    float* imp  = (float*)(ws + OFF_IMP);

    // zero h / c / imp state (deterministic across calls), convert weights to bf16
    (void)hipMemsetAsync(ws + ZERO_OFF, 0, ZERO_BYTES, stream);
    hipLaunchKernelGGL(convert_whh, dim3(8192), dim3(256), 0, stream, Whh_f, Whh_b, Wbf);

    // t = -1: initial cell (x = +/-128, h = c = 0); t = 0..511: scan steps
    for (int t = -1; t < 512; ++t) {
        hipLaunchKernelGGL(lstm_step, dim3(128), dim3(256), 0, stream,
                           Wbf, hbf, hf32, cst, imp, values, masks,
                           Wih_f, bih_f, bhh_f, Wih_b, bih_b, bhh_b,
                           W_imp, b_imp, t);
    }

    hipLaunchKernelGGL(lstm_epilogue, dim3(512), dim3(256), 0, stream,
                       hf32, values, masks, (float*)d_out);
}
